// Attention_46626164965579
// MI455X (gfx1250) — matrix operand order repaired and checked by reading, not compiled, not run on hardware
//
#include <hip/hip_runtime.h>
#include <hip/hip_bf16.h>
#include <stdint.h>

// ---------------------------------------------------------------------------
// Causal MHA forward for MI455X (gfx1250, wave32, WMMA bf16 16x16x32, f32 acc)
// ---------------------------------------------------------------------------

typedef unsigned short ushort_t;
typedef __attribute__((ext_vector_type(16))) __bf16 v16bf;
typedef __attribute__((ext_vector_type(8)))  float  v8f;

union FragB { uint4 u[2]; v16bf v; };   // 32 bytes = 16 bf16 per lane

#define D_MODEL 1024
#define N_HEADS 16
#define HEAD_DIM 64
#define SEQ 2048
#define BATCH 4
#define MTOT (BATCH*SEQ)   // 8192 tokens
#define PSTR 40            // padded LDS stride (bf16 elems) for P tiles

__device__ __forceinline__ ushort_t f2bf(float f) {
  uint32_t u = __float_as_uint(f);
  u += 0x7FFFu + ((u >> 16) & 1u);     // round-to-nearest-even
  return (ushort_t)(u >> 16);
}

// ---------------------------------------------------------------- convert ---
__global__ void cvt_bf16(const float* __restrict__ src,
                         ushort_t* __restrict__ dst, int n4) {
  int i = blockIdx.x * blockDim.x + threadIdx.x;
  if (i < n4) {
    float4 f = reinterpret_cast<const float4*>(src)[i];
    ushort4 o;
    o.x = f2bf(f.x); o.y = f2bf(f.y); o.z = f2bf(f.z); o.w = f2bf(f.w);
    reinterpret_cast<ushort4*>(dst)[i] = o;
  }
}

// ------------------------------------------------------------------- GEMM ---
// C[M,N] = A[M,K] * W[N,K]^T  (torch Linear). A,W bf16 row-major, K=N=1024.
// Block = 256 threads (8 waves) -> 64(M) x 256(N) tile; wave = 32(M) x 64(N).
// mode 0: bf16 out, layout [B,H,S,Hd]    (Q and K projections)
// mode 2: bf16 out, layout [B,H,Hd,S]    (V projection, pre-transposed)
// mode 3: f32 out row-major [M,N] + bias (final projection)
__global__ __launch_bounds__(256)
void gemm_wmma(const ushort_t* __restrict__ A,
               const ushort_t* __restrict__ W,
               void* __restrict__ dst,
               const float* __restrict__ bias,
               int mode)
{
  const int lane = threadIdx.x & 31;
  const int wave = threadIdx.x >> 5;
  const int half = lane >> 4;
  const int lr   = lane & 15;
  const int wy = wave >> 2;                 // 0..1 (M dir)
  const int wx = wave & 3;                  // 0..3 (N dir)
  const int m0 = blockIdx.y * 64  + wy * 32;
  const int n0 = blockIdx.x * 256 + wx * 64;

  v8f acc[2][4];
  const v8f vzero = {0.f,0.f,0.f,0.f,0.f,0.f,0.f,0.f};
  #pragma unroll
  for (int r = 0; r < 2; ++r)
    #pragma unroll
    for (int c = 0; c < 4; ++c) acc[r][c] = vzero;

  for (int kk = 0; kk < D_MODEL; kk += 32) {
    FragB aF[2], bF[4];
    #pragma unroll
    for (int rt = 0; rt < 2; ++rt) {        // A frag: lane row m, k in 2 chunks
      const ushort_t* p = A + (size_t)(m0 + rt*16 + lr) * D_MODEL + kk + half*16;
      aF[rt].u[0] = *reinterpret_cast<const uint4*>(p);
      aF[rt].u[1] = *reinterpret_cast<const uint4*>(p + 8);
    }
    #pragma unroll
    for (int ct = 0; ct < 4; ++ct) {        // B frag: lane col n = W row n
      const ushort_t* p = W + (size_t)(n0 + ct*16 + lr) * D_MODEL + kk + half*16;
      bF[ct].u[0] = *reinterpret_cast<const uint4*>(p);
      bF[ct].u[1] = *reinterpret_cast<const uint4*>(p + 8);
    }
    if (kk + 32 < D_MODEL) {                // global_prefetch_b8 for next A rows
      __builtin_prefetch(A + (size_t)(m0 + lr) * D_MODEL + kk + 32, 0, 1);
      __builtin_prefetch(A + (size_t)(m0 + 16 + lr) * D_MODEL + kk + 32, 0, 1);
    }
    #pragma unroll
    for (int rt = 0; rt < 2; ++rt)
      #pragma unroll
      for (int ct = 0; ct < 4; ++ct)
        acc[rt][ct] = __builtin_amdgcn_wmma_f32_16x16x32_bf16(
            false, aF[rt].v, false, bF[ct].v, (short)0, acc[rt][ct],
            false, false);
  }

  // Epilogue: C layout is row m = v + 8*half, col n = lane%16.
  #pragma unroll
  for (int rt = 0; rt < 2; ++rt) {
    #pragma unroll
    for (int ct = 0; ct < 4; ++ct) {
      const int n = n0 + ct*16 + lr;
      const float bval = (mode == 3) ? bias[n] : 0.0f;
      #pragma unroll
      for (int v = 0; v < 8; ++v) {
        const int m = m0 + rt*16 + v + 8*half;
        const float val = acc[rt][ct][v];
        if (mode == 3) {
          reinterpret_cast<float*>(dst)[(size_t)m * D_MODEL + n] = val + bval;
        } else {
          const int b = m >> 11, s = m & 2047, h = n >> 6, d = n & 63;
          size_t idx;
          if (mode == 2)  // V transposed: [B,H,Hd,S]
            idx = ((size_t)((b << 4) + h) * HEAD_DIM + d) * SEQ + s;
          else            // Q/K: [B,H,S,Hd]
            idx = ((size_t)((b << 4) + h) * SEQ + s) * HEAD_DIM + d;
          reinterpret_cast<ushort_t*>(dst)[idx] = f2bf(val);
        }
      }
    }
  }
}

// -------------------------------------------------------------- attention ---
// One wave = 16 queries of one (b,h). Scores computed transposed (M=key,
// N=query) so softmax over keys is lane-local + one shfl_xor(16).
// P -> LDS (bf16, [q][k], padded) -> B fragment (2x ds_load_b128).
// O accumulated transposed: O^T[d][q] = V^T[d][k] * P^T[k][q].
__global__ __launch_bounds__(128)
void attn_fwd(const ushort_t* __restrict__ Q,    // [BH, SEQ, 64] bf16
              const ushort_t* __restrict__ K,    // [BH, SEQ, 64] bf16
              const ushort_t* __restrict__ Vt,   // [BH, 64, SEQ] bf16
              ushort_t* __restrict__ Ctx)        // [MTOT, 1024]  bf16
{
  __shared__ __attribute__((aligned(16))) ushort_t Pbuf[4 * 16 * PSTR];
  const int lane = threadIdx.x & 31;
  const int wave = threadIdx.x >> 5;
  const int half = lane >> 4;
  const int lr   = lane & 15;

  const int strip = blockIdx.x * 4 + wave;   // 8192 strips total
  const int bh    = strip >> 7;              // 128 strips per (b,h)
  const int qbase = (strip & 127) * 16;
  const int h     = bh & 15;

  const ushort_t* Qb = Q  + (size_t)bh * SEQ * HEAD_DIM;
  const ushort_t* Kb = K  + (size_t)bh * SEQ * HEAD_DIM;
  const ushort_t* Vb = Vt + (size_t)bh * HEAD_DIM * SEQ;
  ushort_t* Pl = Pbuf + wave * 16 * PSTR;

  // Q fragments (B operand of S^T), one per 32-wide d-chunk; loaded once.
  FragB qf[2];
  #pragma unroll
  for (int dc = 0; dc < 2; ++dc) {
    const ushort_t* p = Qb + (size_t)(qbase + lr) * HEAD_DIM + dc*32 + half*16;
    qf[dc].u[0] = *reinterpret_cast<const uint4*>(p);
    qf[dc].u[1] = *reinterpret_cast<const uint4*>(p + 8);
  }

  const v8f vzero = {0.f,0.f,0.f,0.f,0.f,0.f,0.f,0.f};
  v8f o[4];
  #pragma unroll
  for (int dt = 0; dt < 4; ++dt) o[dt] = vzero;
  float mrun = -3.0e38f, lsum = 0.0f;

  const int qidx = qbase + lr;                 // this lane's query index
  const int nblocks = (qbase + 16 + 31) >> 5;  // causal: keys 0..qbase+15

  for (int kb = 0; kb < nblocks; ++kb) {
    const int kbase = kb * 32;

    // --- scores S^T for 32 keys: two 16x16 tiles, K-dim = Hd = 64 (2 wmma)
    v8f sc[2];
    #pragma unroll
    for (int t = 0; t < 2; ++t) {
      v8f a = vzero;
      #pragma unroll
      for (int dc = 0; dc < 2; ++dc) {
        FragB kf;
        const ushort_t* p =
            Kb + (size_t)(kbase + t*16 + lr) * HEAD_DIM + dc*32 + half*16;
        kf.u[0] = *reinterpret_cast<const uint4*>(p);
        kf.u[1] = *reinterpret_cast<const uint4*>(p + 8);
        a = __builtin_amdgcn_wmma_f32_16x16x32_bf16(
            false, kf.v, false, qf[dc].v, (short)0, a, false, false);
      }
      sc[t] = a;
    }

    // --- scale + causal mask + block max (lane-local then pairwise shfl)
    float bm = -3.0e38f;
    #pragma unroll
    for (int t = 0; t < 2; ++t)
      #pragma unroll
      for (int v = 0; v < 8; ++v) {
        const int key = kbase + t*16 + v + 8*half;
        float s = sc[t][v] * 0.125f;             // 1/sqrt(64)
        s = (key <= qidx) ? s : -3.0e38f;
        sc[t][v] = s;
        bm = fmaxf(bm, s);
      }
    bm = fmaxf(bm, __shfl_xor(bm, 16, 32));

    const float mnew  = fmaxf(mrun, bm);
    const float alpha = __expf(mrun - mnew);
    float psum = 0.0f;
    #pragma unroll
    for (int t = 0; t < 2; ++t)
      #pragma unroll
      for (int v = 0; v < 8; ++v) {
        const float p = __expf(sc[t][v] - mnew);
        psum += p;
        Pl[lr * PSTR + t*16 + v + 8*half] = f2bf(p);  // LDS as P[q][k]
      }
    psum += __shfl_xor(psum, 16, 32);
    lsum = lsum * alpha + psum;
    mrun = mnew;
    #pragma unroll
    for (int dt = 0; dt < 4; ++dt)
      #pragma unroll
      for (int v = 0; v < 8; ++v) o[dt][v] *= alpha;

    // --- PV: O^T += V^T * P^T.  B frag of P from LDS (shared by 4 tiles).
    FragB pf;
    {
      const ushort_t* p = Pl + lr * PSTR + half * 16;
      pf.u[0] = *reinterpret_cast<const uint4*>(p);
      pf.u[1] = *reinterpret_cast<const uint4*>(p + 8);
    }
    #pragma unroll
    for (int dt = 0; dt < 4; ++dt) {
      FragB vf;  // A frag: lane row d = dt*16+lr, k = keys (contig in Vt)
      const ushort_t* p = Vb + (size_t)(dt*16 + lr) * SEQ + kbase + half*16;
      vf.u[0] = *reinterpret_cast<const uint4*>(p);
      vf.u[1] = *reinterpret_cast<const uint4*>(p + 8);
      o[dt] = __builtin_amdgcn_wmma_f32_16x16x32_bf16(
          false, vf.v, false, pf.v, (short)0, o[dt], false, false);
    }
  }

  // --- finalize: ctx[token][h*64+d] = O^T / lsum  (bf16 for final GEMM)
  const float inv = 1.0f / lsum;
  const int tok = (bh >> 4) * SEQ + qbase + lr;   // b*SEQ + s
  #pragma unroll
  for (int dt = 0; dt < 4; ++dt)
    #pragma unroll
    for (int v = 0; v < 8; ++v) {
      const int d = dt*16 + v + 8*half;
      Ctx[(size_t)tok * D_MODEL + h * HEAD_DIM + d] = f2bf(o[dt][v] * inv);
    }
}

// ------------------------------------------------------------------ driver ---
extern "C" void kernel_launch(void* const* d_in, const int* in_sizes, int n_in,
                              void* d_out, int out_size, void* d_ws, size_t ws_size,
                              hipStream_t stream) {
  const float* x  = (const float*)d_in[0];
  const float* Wq = (const float*)d_in[1];
  const float* Wk = (const float*)d_in[2];
  const float* Wv = (const float*)d_in[3];
  const float* Wp = (const float*)d_in[4];
  const float* bp = (const float*)d_in[5];
  float* out = (float*)d_out;

  ushort_t* ws = (ushort_t*)d_ws;
  const size_t XB = (size_t)MTOT * D_MODEL;     // 8.39M elems per activation
  const size_t WB = (size_t)D_MODEL * D_MODEL;  // 1.05M elems per weight
  ushort_t* xb  = ws;
  ushort_t* wqb = xb  + XB;
  ushort_t* wkb = wqb + WB;
  ushort_t* wvb = wkb + WB;
  ushort_t* wpb = wvb + WB;
  ushort_t* qb  = wpb + WB;
  ushort_t* kk  = qb  + XB;
  ushort_t* vt  = kk  + XB;
  ushort_t* ctx = vt  + XB;

  const int n4 = (int)(XB / 4), w4 = (int)(WB / 4);
  cvt_bf16<<<(n4 + 255) / 256, 256, 0, stream>>>(x,  xb,  n4);
  cvt_bf16<<<(w4 + 255) / 256, 256, 0, stream>>>(Wq, wqb, w4);
  cvt_bf16<<<(w4 + 255) / 256, 256, 0, stream>>>(Wk, wkb, w4);
  cvt_bf16<<<(w4 + 255) / 256, 256, 0, stream>>>(Wv, wvb, w4);
  cvt_bf16<<<(w4 + 255) / 256, 256, 0, stream>>>(Wp, wpb, w4);

  dim3 g(D_MODEL / 256, MTOT / 64);   // (4, 128) blocks of 256 threads
  gemm_wmma<<<g, 256, 0, stream>>>(xb, wqb, qb, nullptr, 0);   // Q
  gemm_wmma<<<g, 256, 0, stream>>>(xb, wkb, kk, nullptr, 0);   // K
  gemm_wmma<<<g, 256, 0, stream>>>(xb, wvb, vt, nullptr, 2);   // V^T

  const int strips = BATCH * N_HEADS * (SEQ / 16);             // 8192
  attn_fwd<<<strips / 4, 128, 0, stream>>>(qb, kk, vt, ctx);

  gemm_wmma<<<g, 256, 0, stream>>>(ctx, wpb, out, bp, 3);      // out proj + bias
}
